// ExpertChoiceMoE_9431748182293
// MI455X (gfx1250) — matrix operand order repaired and checked by reading, compiled, not run on hardware
//
#include <hip/hip_runtime.h>
#include <hip/hip_bf16.h>
#include <stdint.h>

// Problem constants (from reference)
#define B_TOK 8192
#define D_DIM 1024
#define H_DIM 4096
#define O_DIM 1024
#define E_NUM 16
#define CAP   1024

typedef __attribute__((ext_vector_type(16))) __bf16 v16bf;
typedef __attribute__((ext_vector_type(8)))  __bf16 v8bf;
typedef __attribute__((ext_vector_type(8)))  float  v8f;
typedef __attribute__((ext_vector_type(4)))  unsigned short u16x4;
typedef __attribute__((ext_vector_type(2)))  unsigned short u16x2;

__device__ __forceinline__ unsigned short f2bf_u16(float f) {
  unsigned u = __float_as_uint(f);
  unsigned r = (u + 0x7FFFu + ((u >> 16) & 1u)) >> 16;  // round-to-nearest-even
  return (unsigned short)r;
}

// ---------------------------------------------------------------- zero init
__global__ void moe_zero_f32(float* p, long long n) {
  long long i = (long long)blockIdx.x * blockDim.x + threadIdx.x;
  long long stride = (long long)gridDim.x * blockDim.x;
  for (; i < n; i += stride) p[i] = 0.0f;
}

// ---------------------------------------------------------------- fp32 -> bf16 (row-major copy, for x)
__global__ void moe_cvt_bf16(const float* __restrict__ src,
                             unsigned short* __restrict__ dst, long long n) {
  long long i = ((long long)blockIdx.x * blockDim.x + threadIdx.x) * 4;
  long long stride = (long long)gridDim.x * blockDim.x * 4;
  for (; i < n; i += stride) {
    float4 v = *(const float4*)(src + i);
    u16x4 o;
    o[0] = f2bf_u16(v.x); o[1] = f2bf_u16(v.y);
    o[2] = f2bf_u16(v.z); o[3] = f2bf_u16(v.w);
    *(u16x4*)(dst + i) = o;
  }
}

// ---------------------------------------------------------------- fp32 -> bf16 convert + transpose
// src [E][K][N] f32  ->  dst [E][N][K] bf16   (32x32 LDS tiles)
// K-major weights make the GEMM B tile a contiguous-row copy -> pure async-to-LDS staging.
__global__ __launch_bounds__(256) void moe_cvt_t_bf16(const float* __restrict__ src,
                                                      unsigned short* __restrict__ dst,
                                                      int K, int N) {
  __shared__ unsigned short t[32][33];
  const int e = blockIdx.z;
  const int n0 = blockIdx.x * 32, k0 = blockIdx.y * 32;
  const int tx = threadIdx.x & 31, ty = threadIdx.x >> 5;      // load layout
  const float* s = src + ((long long)e * K + k0) * N + n0;
#pragma unroll
  for (int i = 0; i < 4; ++i) {
    int k = ty + i * 8;
    t[k][tx] = f2bf_u16(s[(long long)k * N + tx]);
  }
  __syncthreads();
  const int tx2 = (threadIdx.x & 15) * 2, ty2 = threadIdx.x >> 4;  // store layout
  unsigned short* d = dst + ((long long)e * N + n0) * K + k0;
#pragma unroll
  for (int i = 0; i < 2; ++i) {
    int n = ty2 + i * 16;
    u16x2 v;
    v[0] = t[tx2][n];
    v[1] = t[tx2 + 1][n];
    *(u16x2*)(d + (long long)n * K + tx2) = v;
  }
}

// ---------------------------------------------------------------- router GEMM (fp32 exact)
__global__ void moe_router(const float* __restrict__ x, const float* __restrict__ Wr,
                           const float* __restrict__ br, float* __restrict__ logits) {
  int t = threadIdx.x >> 4;
  int e = threadIdx.x & 15;
  int token = blockIdx.x * 16 + t;
  const float* xr = x + (long long)token * D_DIM;
  float acc = 0.f;
  for (int d = 0; d < D_DIM; ++d) acc += xr[d] * Wr[d * E_NUM + e];
  logits[token * E_NUM + e] = acc + br[e];
}

// ---------------------------------------------------------------- softmax over tokens (axis 0)
__global__ void moe_softmax_col(const float* __restrict__ logits, float* __restrict__ probs) {
  int e = blockIdx.x;
  __shared__ float red[256];
  float m = -3.4e38f;
  for (int i = threadIdx.x; i < B_TOK; i += 256) m = fmaxf(m, logits[i * E_NUM + e]);
  red[threadIdx.x] = m; __syncthreads();
  for (int s = 128; s > 0; s >>= 1) {
    if (threadIdx.x < s) red[threadIdx.x] = fmaxf(red[threadIdx.x], red[threadIdx.x + s]);
    __syncthreads();
  }
  m = red[0]; __syncthreads();
  float sum = 0.f;
  for (int i = threadIdx.x; i < B_TOK; i += 256) sum += __expf(logits[i * E_NUM + e] - m);
  red[threadIdx.x] = sum; __syncthreads();
  for (int s = 128; s > 0; s >>= 1) {
    if (threadIdx.x < s) red[threadIdx.x] += red[threadIdx.x + s];
    __syncthreads();
  }
  float inv = 1.f / red[0];
  for (int i = threadIdx.x; i < B_TOK; i += 256)
    probs[i * E_NUM + e] = __expf(logits[i * E_NUM + e] - m) * inv;
}

// ---------------------------------------------------------------- top-1024 per expert
__global__ __launch_bounds__(1024) void moe_topk(const float* __restrict__ probs,
                                                 int* __restrict__ top_idx,
                                                 float* __restrict__ top_scores) {
  int e = blockIdx.x;
  __shared__ unsigned long long key[B_TOK];  // 64KB (<=320KB/WG on CDNA5)
  for (int i = threadIdx.x; i < B_TOK; i += 1024) {
    unsigned u = __float_as_uint(probs[i * E_NUM + e]);
    u = (u & 0x80000000u) ? ~u : (u | 0x80000000u);
    key[i] = ((unsigned long long)u << 32) | (unsigned long long)(0xFFFFFFFFu - (unsigned)i);
  }
  __syncthreads();
  for (int k = 2; k <= B_TOK; k <<= 1) {
    for (int j = k >> 1; j > 0; j >>= 1) {
      for (int i = threadIdx.x; i < B_TOK; i += 1024) {
        int ixj = i ^ j;
        if (ixj > i) {
          bool desc = ((i & k) == 0);
          unsigned long long a = key[i], b = key[ixj];
          if ((a < b) == desc) { key[i] = b; key[ixj] = a; }
        }
      }
      __syncthreads();
    }
  }
  if (threadIdx.x < CAP) {
    unsigned long long kk = key[threadIdx.x];
    unsigned idx = 0xFFFFFFFFu - (unsigned)(kk & 0xFFFFFFFFull);
    top_idx[e * CAP + threadIdx.x] = (int)idx;
    top_scores[e * CAP + threadIdx.x] = probs[idx * E_NUM + e];
  }
}

// ---------------------------------------------------------------- expert mask
__global__ void moe_mask(const int* __restrict__ top_idx, float* __restrict__ mask) {
  int g = blockIdx.x * blockDim.x + threadIdx.x;
  int e = g / CAP, j = g - e * CAP;
  int idx = top_idx[e * CAP + j];
  mask[idx * E_NUM + e] = 1.0f;
}

// ---------------------------------------------------------------- WMMA GEMM
// Fragment packing (wave32), contiguous k order shared by A and B:
//  A 16x32 bf16: lane&15 -> M, K = kq..kq+15 (kq=0/16 by lane>>4)
//                -> two contiguous 16B LDS chunks (row-major A tile).
//  B 32x16 bf16: lane&15 = N, K-half by lane>>4; B tile staged [N][K] (weights
//                pre-transposed) -> fragment is 16 contiguous bf16.
//  C/D 16x16 f32: VGPR r -> M=r (lanes 0-15) / M=r+8 (lanes 16-31), N=lane&15.

#define BM 128
#define BN 128
#define KC 32
#define LDA 40          // A tile row stride (bf16): 80B rows, 16B-aligned chunks
#define LDBT 40         // B tile (transposed) column stride (bf16)
#define ABYTES (BM * LDA * 2)
#define BBYTES (BN * LDBT * 2)

__device__ __forceinline__ v16bf frag_cat(v8bf lo, v8bf hi) {
  return __builtin_shufflevector(lo, hi, 0, 1, 2, 3, 4, 5, 6, 7,
                                 8, 9, 10, 11, 12, 13, 14, 15);
}

// gfx1250 GLOBAL_LOAD_ASYNC_TO_LDS_B128 (ASYNCcnt-tracked DMA to LDS)
__device__ __forceinline__ void async_b128(unsigned lds_addr, unsigned voff,
                                           unsigned long long gbase) {
  asm volatile("global_load_async_to_lds_b128 %0, %1, %2"
               :: "v"(lds_addr), "v"(voff), "s"(gbase) : "memory");
}
__device__ __forceinline__ void wait_async_le4() {  // current buffer done, next in flight
  asm volatile("s_wait_asynccnt 0x4" ::: "memory");
}
__device__ __forceinline__ void wait_async_le0() {
  asm volatile("s_wait_asynccnt 0x0" ::: "memory");
}

// GEMM1: h[e, CAP, H] = relu( xbf[top_idx[e]] @ W1t[e]^T + b1[e] )  -> bf16
__global__ __launch_bounds__(256) void moe_gemm1(const unsigned short* __restrict__ xbf,
                                                 const unsigned short* __restrict__ w1t,  // [E][H][D]
                                                 const float* __restrict__ b1,
                                                 const int* __restrict__ top_idx,
                                                 unsigned short* __restrict__ hbuf) {
  const int e = blockIdx.z;
  const int nbase = blockIdx.x * BN;
  const int mbase = blockIdx.y * BM;
  __shared__ __bf16 As[2][BM * LDA];
  __shared__ __bf16 Bs[2][BN * LDBT];
  __shared__ int gidx[BM];
  const int tid = threadIdx.x;
  if (tid < BM) gidx[tid] = top_idx[e * CAP + mbase + tid];
  __syncthreads();

  const int wave = tid >> 5, lane = tid & 31;
  const int wm = (wave >> 2) * 64;
  const int wn = (wave & 3) * 32;
  v8f acc[4][2] = {};

  const int Mr    = lane & 15;
  const int khalf = (lane >> 4) << 3;  // 0 or 8
  const int kq    = (lane >> 4) << 4;  // 0 or 16
  const int Nc    = lane & 15;

  // per-thread constant async addressing: 512 16B chunks per tile, 2/thread
  const int q0 = tid, q1 = tid + 256;
  const int r0 = q0 >> 2, r1 = q1 >> 2;                 // tile row / col index
  const unsigned off0 = (q0 & 3) * 16u, off1 = (q1 & 3) * 16u;
  const unsigned ldsAb = (unsigned)(size_t)(&As[0][0]);
  const unsigned ldsBb = (unsigned)(size_t)(&Bs[0][0]);
  const unsigned ldsA0 = ldsAb + (unsigned)r0 * (LDA * 2) + off0;
  const unsigned ldsA1 = ldsAb + (unsigned)r1 * (LDA * 2) + off1;
  const unsigned ldsB0 = ldsBb + (unsigned)r0 * (LDBT * 2) + off0;
  const unsigned ldsB1 = ldsBb + (unsigned)r1 * (LDBT * 2) + off1;
  const unsigned voffA0 = (unsigned)gidx[r0] * (D_DIM * 2) + off0;   // gathered x rows
  const unsigned voffA1 = (unsigned)gidx[r1] * (D_DIM * 2) + off1;
  const unsigned voffB0 = (unsigned)r0 * (D_DIM * 2) + off0;         // K-major W1t rows
  const unsigned voffB1 = (unsigned)r1 * (D_DIM * 2) + off1;
  const unsigned long long xbase = (unsigned long long)(size_t)xbf;
  const unsigned long long bbase =
      (unsigned long long)(size_t)(w1t + ((long long)e * H_DIM + nbase) * D_DIM);

  auto issue = [&](int buf, int k0) {
    unsigned bo = (unsigned)buf;
    unsigned long long gA = xbase + (unsigned long long)(k0 * 2);
    unsigned long long gB = bbase + (unsigned long long)(k0 * 2);
    async_b128(ldsA0 + bo * ABYTES, voffA0, gA);
    async_b128(ldsA1 + bo * ABYTES, voffA1, gA);
    async_b128(ldsB0 + bo * BBYTES, voffB0, gB);
    async_b128(ldsB1 + bo * BBYTES, voffB1, gB);
  };

  issue(0, 0);
  int p = 0;
  for (int k0 = 0; k0 < D_DIM; k0 += KC) {
    if (k0 + KC < D_DIM) { issue(p ^ 1, k0 + KC); wait_async_le4(); }
    else                 { wait_async_le0(); }
    __syncthreads();

    const __bf16* Ap = &As[p][0];
    const __bf16* Bp = &Bs[p][0];
    v16bf a[4], bfr[2];
#pragma unroll
    for (int mt = 0; mt < 4; ++mt) {
      const v8bf* ap = (const v8bf*)(Ap + (wm + mt * 16 + Mr) * LDA + kq);
      a[mt] = frag_cat(ap[0], ap[1]);
    }
#pragma unroll
    for (int nt = 0; nt < 2; ++nt) {
      const v8bf* bp = (const v8bf*)(Bp + (wn + nt * 16 + Nc) * LDBT + kq);
      bfr[nt] = frag_cat(bp[0], bp[1]);
    }
#pragma unroll
    for (int mt = 0; mt < 4; ++mt)
#pragma unroll
      for (int nt = 0; nt < 2; ++nt)
        acc[mt][nt] = __builtin_amdgcn_wmma_f32_16x16x32_bf16(
            false, a[mt], false, bfr[nt], (short)0, acc[mt][nt], false, false);
    __syncthreads();
    p ^= 1;
  }

#pragma unroll
  for (int mt = 0; mt < 4; ++mt) {
#pragma unroll
    for (int nt = 0; nt < 2; ++nt) {
      int col = nbase + wn + nt * 16 + (lane & 15);
      float bias = b1[e * H_DIM + col];
#pragma unroll
      for (int r = 0; r < 8; ++r) {
        int row = mbase + wm + mt * 16 + ((lane < 16) ? r : r + 8);
        float v = acc[mt][nt][r] + bias;
        v = v > 0.f ? v : 0.f;
        hbuf[((long long)e * CAP + row) * H_DIM + col] = f2bf_u16(v);
      }
    }
  }
}

// GEMM2: out[top_idx[e]] += score * ( h[e] @ W2t[e]^T + b2[e] )
__global__ __launch_bounds__(256) void moe_gemm2(const unsigned short* __restrict__ hbuf,
                                                 const unsigned short* __restrict__ w2t,  // [E][O][H]
                                                 const float* __restrict__ b2,
                                                 const int* __restrict__ top_idx,
                                                 const float* __restrict__ top_scores,
                                                 float* __restrict__ out) {
  const int e = blockIdx.z;
  const int nbase = blockIdx.x * BN;
  const int mbase = blockIdx.y * BM;
  __shared__ __bf16 As[2][BM * LDA];
  __shared__ __bf16 Bs[2][BN * LDBT];
  __shared__ int gidx[BM];
  __shared__ float gsc[BM];
  const int tid = threadIdx.x;
  if (tid < BM) {
    gidx[tid] = top_idx[e * CAP + mbase + tid];
    gsc[tid]  = top_scores[e * CAP + mbase + tid];
  }
  __syncthreads();

  const int wave = tid >> 5, lane = tid & 31;
  const int wm = (wave >> 2) * 64;
  const int wn = (wave & 3) * 32;
  v8f acc[4][2] = {};

  const int Mr    = lane & 15;
  const int khalf = (lane >> 4) << 3;
  const int kq    = (lane >> 4) << 4;
  const int Nc    = lane & 15;

  const int q0 = tid, q1 = tid + 256;
  const int r0 = q0 >> 2, r1 = q1 >> 2;
  const unsigned off0 = (q0 & 3) * 16u, off1 = (q1 & 3) * 16u;
  const unsigned ldsAb = (unsigned)(size_t)(&As[0][0]);
  const unsigned ldsBb = (unsigned)(size_t)(&Bs[0][0]);
  const unsigned ldsA0 = ldsAb + (unsigned)r0 * (LDA * 2) + off0;
  const unsigned ldsA1 = ldsAb + (unsigned)r1 * (LDA * 2) + off1;
  const unsigned ldsB0 = ldsBb + (unsigned)r0 * (LDBT * 2) + off0;
  const unsigned ldsB1 = ldsBb + (unsigned)r1 * (LDBT * 2) + off1;
  const unsigned voffA0 = (unsigned)r0 * (H_DIM * 2) + off0;   // contiguous h rows
  const unsigned voffA1 = (unsigned)r1 * (H_DIM * 2) + off1;
  const unsigned voffB0 = (unsigned)r0 * (H_DIM * 2) + off0;   // K-major W2t rows
  const unsigned voffB1 = (unsigned)r1 * (H_DIM * 2) + off1;
  const unsigned long long hbase =
      (unsigned long long)(size_t)(hbuf + ((long long)e * CAP + mbase) * H_DIM);
  const unsigned long long bbase =
      (unsigned long long)(size_t)(w2t + ((long long)e * O_DIM + nbase) * H_DIM);

  auto issue = [&](int buf, int k0) {
    unsigned bo = (unsigned)buf;
    unsigned long long gA = hbase + (unsigned long long)(k0 * 2);
    unsigned long long gB = bbase + (unsigned long long)(k0 * 2);
    async_b128(ldsA0 + bo * ABYTES, voffA0, gA);
    async_b128(ldsA1 + bo * ABYTES, voffA1, gA);
    async_b128(ldsB0 + bo * BBYTES, voffB0, gB);
    async_b128(ldsB1 + bo * BBYTES, voffB1, gB);
  };

  issue(0, 0);
  int p = 0;
  for (int k0 = 0; k0 < H_DIM; k0 += KC) {
    if (k0 + KC < H_DIM) { issue(p ^ 1, k0 + KC); wait_async_le4(); }
    else                 { wait_async_le0(); }
    __syncthreads();

    const __bf16* Ap = &As[p][0];
    const __bf16* Bp = &Bs[p][0];
    v16bf a[4], bfr[2];
#pragma unroll
    for (int mt = 0; mt < 4; ++mt) {
      const v8bf* ap = (const v8bf*)(Ap + (wm + mt * 16 + Mr) * LDA + kq);
      a[mt] = frag_cat(ap[0], ap[1]);
    }
#pragma unroll
    for (int nt = 0; nt < 2; ++nt) {
      const v8bf* bp = (const v8bf*)(Bp + (wn + nt * 16 + Nc) * LDBT + kq);
      bfr[nt] = frag_cat(bp[0], bp[1]);
    }
#pragma unroll
    for (int mt = 0; mt < 4; ++mt)
#pragma unroll
      for (int nt = 0; nt < 2; ++nt)
        acc[mt][nt] = __builtin_amdgcn_wmma_f32_16x16x32_bf16(
            false, a[mt], false, bfr[nt], (short)0, acc[mt][nt], false, false);
    __syncthreads();
    p ^= 1;
  }

#pragma unroll
  for (int mt = 0; mt < 4; ++mt) {
#pragma unroll
    for (int nt = 0; nt < 2; ++nt) {
      int col = nbase + wn + nt * 16 + (lane & 15);
      float bias = b2[e * O_DIM + col];
#pragma unroll
      for (int r = 0; r < 8; ++r) {
        int rl = wm + mt * 16 + ((lane < 16) ? r : r + 8);
        float v = (acc[mt][nt][r] + bias) * gsc[rl];
        atomicAdd(&out[(long long)gidx[rl] * O_DIM + col], v);
      }
    }
  }
}

// ---------------------------------------------------------------- launch
extern "C" void kernel_launch(void* const* d_in, const int* in_sizes, int n_in,
                              void* d_out, int out_size, void* d_ws, size_t ws_size,
                              hipStream_t stream) {
  const float* x  = (const float*)d_in[0];
  const float* Wr = (const float*)d_in[1];
  const float* br = (const float*)d_in[2];
  const float* W1 = (const float*)d_in[3];
  const float* b1 = (const float*)d_in[4];
  const float* W2 = (const float*)d_in[5];
  const float* b2 = (const float*)d_in[6];

  float* out   = (float*)d_out;                     // [B, O]
  float* probs = out + (long long)B_TOK * O_DIM;    // [B, E]
  float* mask  = probs + (long long)B_TOK * E_NUM;  // [B, E]

  char* ws = (char*)d_ws;
  float* logits = (float*)ws;             ws += (size_t)B_TOK * E_NUM * sizeof(float);
  int*   topidx = (int*)ws;               ws += (size_t)E_NUM * CAP * sizeof(int);
  float* topsc  = (float*)ws;             ws += (size_t)E_NUM * CAP * sizeof(float);
  unsigned short* xbf = (unsigned short*)ws;  ws += (size_t)B_TOK * D_DIM * 2;
  unsigned short* w1t = (unsigned short*)ws;  ws += (size_t)E_NUM * D_DIM * H_DIM * 2;
  unsigned short* w2t = (unsigned short*)ws;  ws += (size_t)E_NUM * H_DIM * O_DIM * 2;
  unsigned short* hbuf = (unsigned short*)ws;  // E*CAP*H bf16

  long long nout = (long long)B_TOK * O_DIM + 2LL * B_TOK * E_NUM;
  moe_zero_f32<<<1024, 256, 0, stream>>>(out, nout);
  moe_cvt_bf16<<<2048, 256, 0, stream>>>(x, xbf, (long long)B_TOK * D_DIM);
  moe_cvt_t_bf16<<<dim3(H_DIM / 32, D_DIM / 32, E_NUM), 256, 0, stream>>>(W1, w1t, D_DIM, H_DIM);
  moe_cvt_t_bf16<<<dim3(O_DIM / 32, H_DIM / 32, E_NUM), 256, 0, stream>>>(W2, w2t, H_DIM, O_DIM);
  moe_router<<<B_TOK / 16, 256, 0, stream>>>(x, Wr, br, logits);
  moe_softmax_col<<<E_NUM, 256, 0, stream>>>(logits, probs);
  moe_topk<<<E_NUM, 1024, 0, stream>>>(probs, topidx, topsc);
  moe_mask<<<(E_NUM * CAP) / 256, 256, 0, stream>>>(topidx, mask);
  moe_gemm1<<<dim3(H_DIM / BN, CAP / BM, E_NUM), 256, 0, stream>>>(xbf, w1t, b1, topidx, hbuf);
  moe_gemm2<<<dim3(O_DIM / BN, CAP / BM, E_NUM), 256, 0, stream>>>(hbuf, w2t, b2, topidx, topsc, out);
}
